// BiLSTM_2491081031886
// MI455X (gfx1250) — matrix operand order repaired and checked by reading, not compiled, not run on hardware
//
#include <hip/hip_runtime.h>

// ---------------------------------------------------------------------------
// Fused LSTM + linear for MI455X (gfx1250, wave32, WMMA).
//   B=2048, T=256, F=H=128, gates=4H=512.
// One workgroup (256 threads = 8 waves) owns a 16-row batch tile and runs the
// full T=256 recurrence with LDS-resident bf16 weights (256 KB of 320 KB/WGP).
// sched_group_barrier directives pipeline DS-reads ahead of the 32 WMMAs per
// timestep (4 independent accumulator chains), so LDS latency overlaps WMMA.
// Ping-pong x/h tiles give a single barrier per timestep; the global load of
// x(t+1) is issued right after the barrier to overlap the WMMA section.
// ---------------------------------------------------------------------------

typedef __attribute__((ext_vector_type(16))) __bf16 v16bf;
typedef __attribute__((ext_vector_type(8)))  __bf16 v8bf;
typedef __attribute__((ext_vector_type(8)))  float  v8f;
typedef __attribute__((ext_vector_type(4)))  float  v4f;

#define B_TOT  2048
#define T_LEN  256
#define F_DIM  128
#define H_DIM  128
#define G4     512      // 4*H
#define BTILE  16
#define NTHREADS 256
#define TILE_E (BTILE * F_DIM)   // 2048 elements per staging tile

#if __has_builtin(__builtin_amdgcn_sched_group_barrier)
#define SGB(mask, n, id) __builtin_amdgcn_sched_group_barrier((mask), (n), (id))
#else
#define SGB(mask, n, id)
#endif
// masks: 0x008 = MFMA/WMMA, 0x100 = DS read

__device__ __forceinline__ float fast_sigmoid(float x) {
    return 1.0f / (1.0f + __expf(-x));
}
__device__ __forceinline__ float fast_tanh(float x) {
    float e = __expf(2.0f * x);
    return (e - 1.0f) / (e + 1.0f);
}

#define WMMA_BF16(A, Bm, C) \
    __builtin_amdgcn_wmma_f32_16x16x32_bf16(false, (A), false, (Bm), (short)0, (C), false, false)

__global__ __launch_bounds__(NTHREADS, 1)
void lstm_fused_wmma(const float* __restrict__ x,      // [B, T, F]
                     const float* __restrict__ w_ih,   // [4H, F]
                     const float* __restrict__ w_hh,   // [4H, H]
                     const float* __restrict__ b_ih,   // [4H]
                     const float* __restrict__ b_hh,   // [4H]
                     const float* __restrict__ w_lin,  // [T*H]
                     const float* __restrict__ b_lin,  // [1]
                     float* __restrict__ out) {        // [B]
    extern __shared__ __align__(16) char smem[];
    __bf16* wih  = (__bf16*)smem;                 // 512*128 bf16 = 128 KB
    __bf16* whh  = wih + G4 * F_DIM;              // 512*128 bf16 = 128 KB
    __bf16* xbuf = whh + G4 * H_DIM;              // 2 x 16*128 bf16 = 8 KB
    __bf16* hbuf = xbuf + 2 * TILE_E;             // 2 x 16*128 bf16 = 8 KB
    float*  bias = (float*)(hbuf + 2 * TILE_E);   // 512 f32
    float*  outb = bias + G4;                     // 16  f32

    const int tid    = threadIdx.x;
    const int lane   = tid & 31;
    const int wave   = tid >> 5;        // 0..7
    const int half   = lane >> 4;       // 0/1  (lane group)
    const int nlo    = lane & 15;       // column within 16-wide tile
    const int j      = wave * 16 + nlo; // hidden unit owned by this lane
    const int wgBase = blockIdx.x * BTILE;

    // ---- one-time: convert weights f32 -> bf16 into LDS (coalesced) -------
    #pragma unroll 4
    for (int i = 0; i < 64; ++i) {
        int e4 = i * NTHREADS + tid;           // float4 index, 16384 total
        v4f a = ((const v4f*)w_ih)[e4];
        v4f b = ((const v4f*)w_hh)[e4];
        int e = e4 * 4;
        wih[e + 0] = (__bf16)a.x; wih[e + 1] = (__bf16)a.y;
        wih[e + 2] = (__bf16)a.z; wih[e + 3] = (__bf16)a.w;
        whh[e + 0] = (__bf16)b.x; whh[e + 1] = (__bf16)b.y;
        whh[e + 2] = (__bf16)b.z; whh[e + 3] = (__bf16)b.w;
    }
    for (int i = tid; i < G4; i += NTHREADS) bias[i] = b_ih[i] + b_hh[i];
    for (int i = tid; i < 2 * TILE_E; i += NTHREADS) hbuf[i] = (__bf16)0.0f;
    if (tid < BTILE) outb[tid] = 0.0f;

    // gate biases for this lane's hidden unit (constant over time)
    const float bi = b_ih[0 * H_DIM + j] + b_hh[0 * H_DIM + j];
    const float bf = b_ih[1 * H_DIM + j] + b_hh[1 * H_DIM + j];
    const float bg = b_ih[2 * H_DIM + j] + b_hh[2 * H_DIM + j];
    const float bo = b_ih[3 * H_DIM + j] + b_hh[3 * H_DIM + j];

    v8f cst  = {};  // cell state, rows m = r + 8*half, column j
    v8f pacc = {};  // running w_lin . h partial sums

    // x loader mapping: row m = tid>>4 (0..15), 8 contiguous floats at (tid&15)*8
    const int    xm   = tid >> 4;
    const int    xf   = (tid & 15) * 8;
    const float* xrow = x + ((size_t)(wgBase + xm) * T_LEN) * F_DIM + xf;
    const int    xb   = xm * F_DIM + xf;

    // B-fragment base pointers (gate gi row = gi*128 + j; stride 128*F_DIM elems)
    const __bf16* wiB = wih + (size_t)j * F_DIM + half * 16;
    const __bf16* whB = whh + (size_t)j * H_DIM + half * 16;
    const int GS = 128 * F_DIM;   // 16384 elements between gate blocks

    // software pipeline: x(0) in registers before the loop
    v4f xr0 = ((const v4f*)xrow)[0];
    v4f xr1 = ((const v4f*)xrow)[1];

    for (int t = 0; t < T_LEN; ++t) {
        __bf16*       xt  = xbuf + (t & 1) * TILE_E;         // stage x_t here
        const __bf16* htr = hbuf + ((t + 1) & 1) * TILE_E;   // h from step t-1
        __bf16*       htw = hbuf + (t & 1) * TILE_E;         // h produced now

        // ---- stage x_t tile into LDS as bf16 ------------------------------
        xt[xb + 0] = (__bf16)xr0.x; xt[xb + 1] = (__bf16)xr0.y;
        xt[xb + 2] = (__bf16)xr0.z; xt[xb + 3] = (__bf16)xr0.w;
        xt[xb + 4] = (__bf16)xr1.x; xt[xb + 5] = (__bf16)xr1.y;
        xt[xb + 6] = (__bf16)xr1.z; xt[xb + 7] = (__bf16)xr1.w;
        __syncthreads();   // single barrier per step (ping-pong buffers)

        // ---- issue next x load now: latency hides under the WMMA section --
        if (t + 1 < T_LEN) {
            const float* p = xrow + (size_t)(t + 1) * F_DIM;
            xr0 = ((const v4f*)p)[0];
            xr1 = ((const v4f*)p)[1];
            __builtin_prefetch(p + F_DIM, 0, 0);   // warm L2 for t+2
        }

        // ---- A fragments (16x32 bf16): lane holds row m=lane&15 -----------
        v16bf ax[4], ah[4];
        {
            const int m = lane & 15;
            #pragma unroll
            for (int kt = 0; kt < 4; ++kt) {
                const int kb = kt * 32 + half * 16;
                v8bf xlo = *(const v8bf*)(xt + m * F_DIM + kb);
                v8bf xhi = *(const v8bf*)(xt + m * F_DIM + kb + 8);
                v8bf hlo = *(const v8bf*)(htr + m * H_DIM + kb);
                v8bf hhi = *(const v8bf*)(htr + m * H_DIM + kb + 8);
                #pragma unroll
                for (int i = 0; i < 8; ++i) {
                    ax[kt][i] = xlo[i]; ax[kt][i + 8] = xhi[i];
                    ah[kt][i] = hlo[i]; ah[kt][i + 8] = hhi[i];
                }
            }
        }

        // ---- kt-major: 8 distinct B frags then 8 WMMAs on 4 indep chains --
        v8f acc0 = {}, acc1 = {}, acc2 = {}, acc3 = {};
        #pragma unroll
        for (int kt = 0; kt < 4; ++kt) {
            const int o = kt * 32;
            v16bf bI0 = *(const v16bf*)(wiB + 0 * GS + o);
            v16bf bI1 = *(const v16bf*)(wiB + 1 * GS + o);
            v16bf bI2 = *(const v16bf*)(wiB + 2 * GS + o);
            v16bf bI3 = *(const v16bf*)(wiB + 3 * GS + o);
            v16bf bH0 = *(const v16bf*)(whB + 0 * GS + o);
            v16bf bH1 = *(const v16bf*)(whB + 1 * GS + o);
            v16bf bH2 = *(const v16bf*)(whB + 2 * GS + o);
            v16bf bH3 = *(const v16bf*)(whB + 3 * GS + o);
            acc0 = WMMA_BF16(ax[kt], bI0, acc0);
            acc1 = WMMA_BF16(ax[kt], bI1, acc1);
            acc2 = WMMA_BF16(ax[kt], bI2, acc2);
            acc3 = WMMA_BF16(ax[kt], bI3, acc3);
            acc0 = WMMA_BF16(ah[kt], bH0, acc0);
            acc1 = WMMA_BF16(ah[kt], bH1, acc1);
            acc2 = WMMA_BF16(ah[kt], bH2, acc2);
            acc3 = WMMA_BF16(ah[kt], bH3, acc3);
        }

        // ---- impose DS-read / WMMA software pipeline on this region -------
        // Totals in this block: 80 DS reads (16 A + 64 B halves), 32 WMMAs.
        // Front-load 24 DS reads, then (2 WMMA, 4 DS) x14, then 4 WMMAs.
        SGB(0x100, 24, 0);
        #pragma unroll
        for (int i = 0; i < 14; ++i) {
            SGB(0x008, 2, 0);
            SGB(0x100, 4, 0);
        }
        SGB(0x008, 4, 0);

        // ---- pointwise LSTM cell update + fused linear --------------------
        // (writes go to the *other* h buffer; no second barrier needed)
        const float wl = w_lin[t * H_DIM + j];
        #pragma unroll
        for (int r = 0; r < 8; ++r) {
            float iv = fast_sigmoid(acc0[r] + bi);
            float fv = fast_sigmoid(acc1[r] + bf);
            float gv = fast_tanh   (acc2[r] + bg);
            float ov = fast_sigmoid(acc3[r] + bo);
            float cn = fv * cst[r] + iv * gv;
            cst[r] = cn;
            float hv = ov * fast_tanh(cn);
            htw[(r + half * 8) * H_DIM + j] = (__bf16)hv;  // next step's A
            pacc[r] += wl * hv;                            // fused final linear
        }
    }

    // ---- reduce w_lin.h partials across hidden units (LDS float atomics) --
    __syncthreads();
    #pragma unroll
    for (int r = 0; r < 8; ++r)
        atomicAdd(&outb[r + half * 8], pacc[r]);           // ds_add_f32
    __syncthreads();
    if (tid < BTILE)
        out[wgBase + tid] = outb[tid] + b_lin[0];
}

extern "C" void kernel_launch(void* const* d_in, const int* in_sizes, int n_in,
                              void* d_out, int out_size, void* d_ws, size_t ws_size,
                              hipStream_t stream) {
    (void)in_sizes; (void)n_in; (void)out_size; (void)d_ws; (void)ws_size;
    const float* x     = (const float*)d_in[0];
    const float* w_ih  = (const float*)d_in[1];
    const float* w_hh  = (const float*)d_in[2];
    const float* b_ih  = (const float*)d_in[3];
    const float* b_hh  = (const float*)d_in[4];
    const float* w_lin = (const float*)d_in[5];
    const float* b_lin = (const float*)d_in[6];
    float* out = (float*)d_out;

    size_t shmem = (size_t)(G4 * F_DIM + G4 * H_DIM + 4 * TILE_E) * sizeof(__bf16)
                 + (size_t)(G4 + BTILE) * sizeof(float);   // ~274 KB (< 320 KB/WGP)

    hipFuncSetAttribute((const void*)lstm_fused_wmma,
                        hipFuncAttributeMaxDynamicSharedMemorySize, (int)shmem);

    lstm_fused_wmma<<<B_TOT / BTILE, NTHREADS, shmem, stream>>>(
        x, w_ih, w_hh, b_ih, b_hh, w_lin, b_lin, out);
}
